// TransLayer_54528904790355
// MI455X (gfx1250) — compile-verified, matrix operand order checked by reading, not run on hardware
//
#include <hip/hip_runtime.h>

// ---------------------------------------------------------------------------
// CDNA5 (gfx1250) types & WMMA helpers
// ---------------------------------------------------------------------------
typedef __bf16 bf16;
typedef __attribute__((ext_vector_type(16))) __bf16 bf16x16;
typedef __attribute__((ext_vector_type(8)))  float  f32x8;

// Load one 16x32 bf16 WMMA operand fragment (A layout; identical for B when B
// is supplied transposed, i.e. Bt[n][k]).  Per ISA 7.12.2:
//   lane 0-15  -> row = lane,    holds K = 0..7  (VGPR0..3) and K = 16..23 (VGPR4..7)
//   lane 16-31 -> row = lane-16, holds K = 8..15 and K = 24..31
__device__ __forceinline__ bf16x16 load_frag(const bf16* base, int ld) {
  int lane = threadIdx.x & 31;
  int row  = lane & 15;
  int kh   = (lane >> 4) << 3;                 // 0 or 8
  const bf16* p = base + (size_t)row * ld + kh;
  bf16x16 f;
#pragma unroll
  for (int i = 0; i < 8; ++i) f[i]     = p[i];       // K = kh + i
#pragma unroll
  for (int i = 0; i < 8; ++i) f[i + 8] = p[16 + i];  // K = 16 + kh + i
  return f;
}

__device__ __forceinline__ f32x8 wmma_bf16(bf16x16 a, bf16x16 b, f32x8 c) {
  return __builtin_amdgcn_wmma_f32_16x16x32_bf16(false, a, false, b,
                                                 (short)0, c, false, false);
}

__device__ __forceinline__ int c_col(int lane)  { return lane & 15; }
__device__ __forceinline__ int c_row0(int lane) { return (lane >> 4) << 3; }

// ---------------------------------------------------------------------------
// Async global -> LDS copy (CDNA5 GLOBAL_LOAD_ASYNC_TO_LDS_B128, ASYNCcnt).
// LDS byte offset comes from an addrspace(3) cast (AS3 pointers are offsets
// from the wave's LDS base, which is what the VDST operand expects).
// ---------------------------------------------------------------------------
__device__ __forceinline__ unsigned lds_u32(const void* p) {
  __attribute__((address_space(3))) const void* lp =
      (__attribute__((address_space(3))) const void*)p;
  return (unsigned)(unsigned long long)lp;
}
__device__ __forceinline__ void async_ld16(const void* lds_dst, const void* gsrc) {
  asm volatile("global_load_async_to_lds_b128 %0, %1, off"
               :: "v"(lds_u32(lds_dst)), "v"(gsrc) : "memory");
}
__device__ __forceinline__ void async_wait0() {
  asm volatile("s_wait_asynccnt 0x0" ::: "memory");
}

// ---------------------------------------------------------------------------
// Problem constants
// ---------------------------------------------------------------------------
#define BATCH 4
#define NSEQ  8192
#define DMODEL 512
#define HEADS 8
#define DH    64
#define MLAND 256
#define LWIN  32
#define KER   33
#define BH    (BATCH*HEADS)
#define ROWS  (BATCH*NSEQ)      // 32768

// ---------------------------------------------------------------------------
// 1. LayerNorm (f32 in -> bf16 out).  One 128-thread block per row of 512.
// ---------------------------------------------------------------------------
__global__ __launch_bounds__(128)
void ln_kernel(const float* __restrict__ x, const float* __restrict__ g,
               const float* __restrict__ be, bf16* __restrict__ xn) {
  int row = blockIdx.x;
  int t   = threadIdx.x;
  const float* xr = x + (size_t)row * DMODEL;
  float v[4], s = 0.f, s2 = 0.f;
#pragma unroll
  for (int i = 0; i < 4; ++i) { v[i] = xr[t + 128*i]; s += v[i]; s2 += v[i]*v[i]; }
  __shared__ float r1[128], r2[128];
  r1[t] = s; r2[t] = s2; __syncthreads();
  for (int o = 64; o > 0; o >>= 1) {
    if (t < o) { r1[t] += r1[t+o]; r2[t] += r2[t+o]; }
    __syncthreads();
  }
  float mu   = r1[0] * (1.f/DMODEL);
  float var  = r2[0] * (1.f/DMODEL) - mu*mu;
  float rstd = rsqrtf(var + 1e-5f);
#pragma unroll
  for (int i = 0; i < 4; ++i) {
    int c = t + 128*i;
    xn[(size_t)row*DMODEL + c] = (bf16)((v[i]-mu)*rstd*g[c] + be[c]);
  }
}

// ---------------------------------------------------------------------------
// 2. f32 (RxC) -> bf16 transposed (CxR).  Used for w_qkv^T, w_out^T.
// ---------------------------------------------------------------------------
__global__ __launch_bounds__(256)
void cvtT_kernel(const float* __restrict__ src, bf16* __restrict__ dst,
                 int R, int C) {
  size_t i = (size_t)blockIdx.x*256 + threadIdx.x;
  if (i >= (size_t)R*C) return;
  int r = (int)(i / C), c = (int)(i % C);
  dst[(size_t)c*R + r] = (bf16)src[i];
}

// ---------------------------------------------------------------------------
// 3. QKV GEMM: [32768,512]bf16 @ [512,1536] -> scatter head-major q/k/v bf16.
//    q gets the DH^-0.5 = 0.125 scale.  64x64 tile per 256-thread block.
// ---------------------------------------------------------------------------
__global__ __launch_bounds__(256)
void qkv_kernel(const bf16* __restrict__ xn, const bf16* __restrict__ wT,
                bf16* __restrict__ qb, bf16* __restrict__ kb,
                bf16* __restrict__ vb) {
  int wv = threadIdx.x >> 5, lane = threadIdx.x & 31;
#pragma unroll
  for (int s = 0; s < 2; ++s) {
    int t  = wv*2 + s;
    int m0 = blockIdx.y*64 + (t>>2)*16;
    int n0 = blockIdx.x*64 + (t&3)*16;
    f32x8 acc = {};
    for (int k = 0; k < DMODEL; k += 32)
      acc = wmma_bf16(load_frag(xn + (size_t)m0*DMODEL + k, DMODEL),
                      load_frag(wT + (size_t)n0*DMODEL + k, DMODEL), acc);
    int n   = n0 + c_col(lane);
    int mb  = m0 + c_row0(lane);
    int sec = n / DMODEL;            // 0=q 1=k 2=v
    int cc  = n % DMODEL;
    int h   = cc >> 6, dh = cc & 63;
    float scale = (sec == 0) ? 0.125f : 1.f;
    bf16* dst = (sec == 0) ? qb : (sec == 1 ? kb : vb);
#pragma unroll
    for (int r = 0; r < 8; ++r) {
      int m  = mb + r;
      int b_ = m >> 13;              // /NSEQ
      int nn = m & (NSEQ-1);
      dst[((((size_t)b_*HEADS + h)*NSEQ + nn) << 6) + dh] = (bf16)(acc[r]*scale);
    }
  }
}

// ---------------------------------------------------------------------------
// 4. Landmark mean over 32 consecutive tokens.
// ---------------------------------------------------------------------------
__global__ __launch_bounds__(256)
void landmark_kernel(const bf16* __restrict__ src, bf16* __restrict__ dst) {
  size_t idx = (size_t)blockIdx.x*256 + threadIdx.x;   // BH*MLAND*DH
  int dh = idx & 63;
  int m  = (int)((idx >> 6) & (MLAND-1));
  int bh = (int)(idx >> 14);
  const bf16* p = src + ((size_t)bh*NSEQ + (size_t)m*LWIN)*DH + dh;
  float s = 0.f;
#pragma unroll 8
  for (int i = 0; i < LWIN; ++i) s += (float)p[(size_t)i*DH];
  dst[idx] = (bf16)(s * (1.f/LWIN));
}

// ---------------------------------------------------------------------------
// 5. Generic batched WMMA GEMM: acc = A(MxK) @ Bt(NxK)^T
//    D0 = a0*acc + d0*I (bf16 and/or f32; optionally transposed store)
//    D1 = a1*acc + d1*I (bf16).  Grid: (N/64, M/64, batch).
// ---------------------------------------------------------------------------
__global__ __launch_bounds__(256)
void gemm_kernel(const bf16* __restrict__ A, size_t sA, int lda,
                 const bf16* __restrict__ Bt, size_t sB, int ldb, int K,
                 float a0, float d0, bf16* __restrict__ D0,
                 float* __restrict__ D0f, size_t sD, int ldd, int d0t,
                 float a1, float d1, bf16* __restrict__ D1) {
  int batch = blockIdx.z;
  const bf16* Ab = A  + (size_t)batch*sA;
  const bf16* Bb = Bt + (size_t)batch*sB;
  int wv = threadIdx.x >> 5, lane = threadIdx.x & 31;
#pragma unroll
  for (int s = 0; s < 2; ++s) {
    int t  = wv*2 + s;
    int m0 = blockIdx.y*64 + (t>>2)*16;
    int n0 = blockIdx.x*64 + (t&3)*16;
    f32x8 acc = {};
    for (int k = 0; k < K; k += 32)
      acc = wmma_bf16(load_frag(Ab + (size_t)m0*lda + k, lda),
                      load_frag(Bb + (size_t)n0*ldb + k, ldb), acc);
    int n  = n0 + c_col(lane);
    int mb = m0 + c_row0(lane);
#pragma unroll
    for (int r = 0; r < 8; ++r) {
      int m = mb + r;
      float v = acc[r];
      if (D0 || D0f) {
        float o = a0*v + ((m == n) ? d0 : 0.f);
        size_t off = (size_t)batch*sD +
                     (d0t ? ((size_t)n*ldd + m) : ((size_t)m*ldd + n));
        if (D0)  D0[off]  = (bf16)o;
        if (D0f) D0f[off] = o;
      }
      if (D1) {
        float o = a1*v + ((m == n) ? d1 : 0.f);
        D1[(size_t)batch*sD + (size_t)m*ldd + n] = (bf16)o;
      }
    }
  }
}

// ---------------------------------------------------------------------------
// 6. Row softmax over 256 (attn2).  One wave per row.
// ---------------------------------------------------------------------------
__global__ __launch_bounds__(32)
void softmax256_kernel(const float* __restrict__ S, bf16* __restrict__ P) {
  size_t row = blockIdx.x;
  int lane = threadIdx.x;
  const float* sr = S + row*MLAND;
  float v[8], mx = -1e30f;
#pragma unroll
  for (int i = 0; i < 8; ++i) { v[i] = sr[lane + 32*i]; mx = fmaxf(mx, v[i]); }
  for (int o = 16; o > 0; o >>= 1) mx = fmaxf(mx, __shfl_xor(mx, o, 32));
  float sm = 0.f;
#pragma unroll
  for (int i = 0; i < 8; ++i) { v[i] = __expf(v[i]-mx); sm += v[i]; }
  for (int o = 16; o > 0; o >>= 1) sm += __shfl_xor(sm, o, 32);
  float inv = 1.f / sm;
#pragma unroll
  for (int i = 0; i < 8; ++i) P[row*MLAND + lane + 32*i] = (bf16)(v[i]*inv);
}

// ---------------------------------------------------------------------------
// 7. denom: softmax row-sums are exactly 1 -> denom = global max column-sum.
// ---------------------------------------------------------------------------
__global__ __launch_bounds__(256)
void colsummax_kernel(const bf16* __restrict__ P, float* __restrict__ out) {
  int bh = blockIdx.x, c = threadIdx.x;
  const bf16* p = P + (size_t)bh*MLAND*MLAND + c;
  float s = 0.f;
  for (int r = 0; r < MLAND; ++r) s += (float)p[(size_t)r*MLAND];
  __shared__ float red[256];
  red[c] = s; __syncthreads();
  for (int o = 128; o > 0; o >>= 1) {
    if (c < o) red[c] = fmaxf(red[c], red[c+o]);
    __syncthreads();
  }
  if (c == 0) out[bh] = red[0];
}

__global__ void reducemax_kernel(const float* __restrict__ in,
                                 float* __restrict__ out) {
  if (threadIdx.x == 0) {
    float m = in[0];
    for (int i = 1; i < BH; ++i) m = fmaxf(m, in[i]);
    out[0] = m;
  }
}

// ---------------------------------------------------------------------------
// 8. z0 = attn2^T / denom ;  batched bf16 256x256 transpose.
// ---------------------------------------------------------------------------
__global__ __launch_bounds__(256)
void initz_kernel(const bf16* __restrict__ A2, const float* __restrict__ denom,
                  bf16* __restrict__ Z) {
  size_t i = (size_t)blockIdx.x*256 + threadIdx.x;
  float inv = 1.f / denom[0];
  size_t bh = i >> 16; int rc = (int)(i & 65535), r = rc >> 8, c = rc & 255;
  Z[(bh << 16) + ((size_t)c << 8) + r] = (bf16)((float)A2[i] * inv);
}

__global__ __launch_bounds__(256)
void trans256_kernel(const bf16* __restrict__ S, bf16* __restrict__ D) {
  size_t i = (size_t)blockIdx.x*256 + threadIdx.x;
  size_t bh = i >> 16; int rc = (int)(i & 65535), r = rc >> 8, c = rc & 255;
  D[(bh << 16) + ((size_t)c << 8) + r] = S[i];
}

// ---------------------------------------------------------------------------
// 9. attn3@v flash kernel: av^T[bh][dh][m] = (softmax_n(q_l . k^T) @ v)^T.
//    Grid (M/64, BH); 256 threads; online softmax over 8192 keys, 64/tile.
//    Q and K tiles staged via GLOBAL_LOAD_ASYNC_TO_LDS_B128 (ASYNCcnt).
// ---------------------------------------------------------------------------
__global__ __launch_bounds__(256)
void attn3v_kernel(const bf16* __restrict__ ql, const bf16* __restrict__ kb,
                   const bf16* __restrict__ vb, bf16* __restrict__ avT) {
  int bh = blockIdx.y, m0 = blockIdx.x*64;
  int tid = threadIdx.x, wv = tid >> 5, lane = tid & 31;
  __shared__ __align__(16) bf16 Qs[64*64];
  __shared__ __align__(16) bf16 Ks[64*64];
  __shared__ __align__(16) bf16 VTs[64*64];
  __shared__ __align__(16) bf16 Ps[64*64];
  __shared__ float Ss[64*64];
  __shared__ float rmax[64], rsum[64], rscale[64];

  // async-stage the whole Q block (64x64, contiguous): 512 x 16B chunks
  const bf16* Qg = ql + ((size_t)bh*MLAND + m0)*DH;
#pragma unroll
  for (int c = 0; c < 2; ++c) {
    int idx = tid + c*256;
    async_ld16(Qs + idx*8, Qg + idx*8);
  }
  if (tid < 64) { rmax[tid] = -1e30f; rsum[tid] = 0.f; }

  f32x8 o[2];
#pragma unroll
  for (int s = 0; s < 2; ++s)
#pragma unroll
    for (int r = 0; r < 8; ++r) o[s][r] = 0.f;

  const bf16* Kb = kb + (size_t)bh*NSEQ*DH;
  const bf16* Vb = vb + (size_t)bh*NSEQ*DH;

  for (int kt = 0; kt < NSEQ; kt += 64) {
    // K tile (row-major, contiguous 8KB): async copy to LDS
#pragma unroll
    for (int c = 0; c < 2; ++c) {
      int idx = tid + c*256;
      async_ld16(Ks + idx*8, Kb + (size_t)kt*DH + idx*8);
    }
    // V tile: register path with transpose into VT[d][key]
    for (int i = tid; i < 64*64; i += 256) {
      int r = i >> 6, c = i & 63;
      VTs[(c << 6) | r] = Vb[(size_t)(kt + r)*DH + c];
    }
    async_wait0();
    __syncthreads();
    // S = Q @ K^T  (Bt for Q@K^T is K itself: Bt[key][d])
#pragma unroll
    for (int s = 0; s < 2; ++s) {
      int t = wv*2 + s, mi = t >> 2, ni = t & 3;
      f32x8 a = {};
      for (int k = 0; k < DH; k += 32)
        a = wmma_bf16(load_frag(Qs + mi*16*64 + k, 64),
                      load_frag(Ks + ni*16*64 + k, 64), a);
      int n = ni*16 + c_col(lane), mb = mi*16 + c_row0(lane);
#pragma unroll
      for (int r = 0; r < 8; ++r) Ss[(mb+r)*64 + n] = a[r];
    }
    __syncthreads();
    if (tid < 64) {                            // online softmax bookkeeping
      float* sr = Ss + tid*64;
      float mx = rmax[tid];
      for (int j = 0; j < 64; ++j) mx = fmaxf(mx, sr[j]);
      float sc = __expf(rmax[tid] - mx), sm = 0.f;
      for (int j = 0; j < 64; ++j) {
        float p = __expf(sr[j] - mx);
        Ps[tid*64 + j] = (bf16)p; sm += p;
      }
      rsum[tid] = rsum[tid]*sc + sm;
      rscale[tid] = sc; rmax[tid] = mx;
    }
    __syncthreads();
#pragma unroll
    for (int s = 0; s < 2; ++s) {              // rescale O, O += P @ V
      int t = wv*2 + s, mi = t >> 2, ni = t & 3;
      int mb = mi*16 + c_row0(lane);
#pragma unroll
      for (int r = 0; r < 8; ++r) o[s][r] *= rscale[mb+r];
      for (int k = 0; k < 64; k += 32)
        o[s] = wmma_bf16(load_frag(Ps  + mi*16*64 + k, 64),
                         load_frag(VTs + ni*16*64 + k, 64), o[s]);
    }
    __syncthreads();
  }
#pragma unroll
  for (int s = 0; s < 2; ++s) {                // write av^T, normalized
    int t = wv*2 + s, mi = t >> 2, ni = t & 3;
    int n = ni*16 + c_col(lane);
    int mloc = mi*16 + c_row0(lane);
#pragma unroll
    for (int r = 0; r < 8; ++r)
      avT[((size_t)bh*DH + n)*MLAND + m0 + mloc + r] =
          (bf16)(o[s][r] / rsum[mloc + r]);
  }
}

// ---------------------------------------------------------------------------
// 10. Depthwise conv (kernel 33 over sequence) -> ao (f32, merged heads).
// ---------------------------------------------------------------------------
__global__ __launch_bounds__(256)
void conv_kernel(const bf16* __restrict__ vb, const float* __restrict__ ker,
                 float* __restrict__ ao) {
  size_t i = (size_t)blockIdx.x*256 + threadIdx.x;  // BH*NSEQ*DH
  int dh = (int)(i & 63);
  int n  = (int)((i >> 6) & (NSEQ-1));
  int bh = (int)(i >> 19);
  int b = bh >> 3, h = bh & 7;
  const bf16* v = vb + (size_t)bh*NSEQ*DH + dh;
  int np = (n + 48 < NSEQ) ? n + 48 : NSEQ - 1;     // prefetch ahead of window
  __builtin_prefetch(v + (size_t)np*DH, 0, 1);
  float s = 0.f;
#pragma unroll
  for (int j = 0; j < KER; ++j) {
    int p = n + j - KER/2;
    if (p >= 0 && p < NSEQ) s += ker[h*KER + j] * (float)v[(size_t)p*DH];
  }
  ao[((size_t)b*NSEQ + n)*DMODEL + h*DH + dh] = s;
}

// ---------------------------------------------------------------------------
// 11. attn1 stage: ao += softmax(q . k_l^T) @ W  (256 keys, all in LDS).
//     Grid (N/32, BH); async-to-LDS staging for Q, k_l chunks, W^T chunks.
// ---------------------------------------------------------------------------
__global__ __launch_bounds__(256)
void attn1_kernel(const bf16* __restrict__ qb, const bf16* __restrict__ kl,
                  const bf16* __restrict__ Wt, float* __restrict__ ao) {
  int bh = blockIdx.y, n0 = blockIdx.x*32;
  int tid = threadIdx.x, wv = tid >> 5, lane = tid & 31;
  int b = bh >> 3, h = bh & 7;
  __shared__ __align__(16) bf16 Qs[32*64];
  __shared__ __align__(16) bf16 Cs[64*64];
  __shared__ __align__(16) bf16 Ps[32*256];
  __shared__ float Ss[32*256];
  __shared__ float rowinv[32];

  const bf16* Q = qb + ((size_t)bh*NSEQ + n0)*DH;
  async_ld16(Qs + tid*8, Q + tid*8);           // 32x64 = 256 x 16B chunks

  for (int kc = 0; kc < 4; ++kc) {             // scores: 32x256 in 64-chunks
    __syncthreads();
    const bf16* Kg = kl + ((size_t)bh*MLAND + kc*64)*DH;   // contiguous 8KB
#pragma unroll
    for (int c = 0; c < 2; ++c) {
      int idx = tid + c*256;
      async_ld16(Cs + idx*8, Kg + idx*8);
    }
    async_wait0();
    __syncthreads();
    int mi = wv >> 2, ni = wv & 3;
    f32x8 a = {};
    for (int k = 0; k < DH; k += 32)
      a = wmma_bf16(load_frag(Qs + mi*16*64 + k, 64),
                    load_frag(Cs + ni*16*64 + k, 64), a);
    int n = kc*64 + ni*16 + c_col(lane);
    int mb = mi*16 + c_row0(lane);
#pragma unroll
    for (int r = 0; r < 8; ++r) Ss[(mb+r)*256 + n] = a[r];
  }
  __syncthreads();
  if (tid < 32) {                              // softmax over 256
    float* sr = Ss + tid*256;
    float mx = -1e30f;
    for (int j = 0; j < 256; ++j) mx = fmaxf(mx, sr[j]);
    float sm = 0.f;
    for (int j = 0; j < 256; ++j) {
      float p = __expf(sr[j] - mx);
      Ps[tid*256 + j] = (bf16)p; sm += p;
    }
    rowinv[tid] = 1.f / sm;
  }
  __syncthreads();
  int mi = wv >> 2, ni = wv & 3;               // O = P @ W (Bt = Wt[dh][m])
  f32x8 o = {};
  for (int kc = 0; kc < 4; ++kc) {
    // Wt chunk: rows d=0..63, cols kc*64..kc*64+63 (128B row segments)
#pragma unroll
    for (int c = 0; c < 2; ++c) {
      int idx = tid + c*256;                   // 512 x 16B chunks
      int rr = idx >> 3, c8 = idx & 7;
      async_ld16(Cs + idx*8,
                 Wt + ((size_t)bh*DH + rr)*MLAND + kc*64 + c8*8);
    }
    async_wait0();
    __syncthreads();
    for (int k = 0; k < 64; k += 32)
      o = wmma_bf16(load_frag(Ps + mi*16*256 + kc*64 + k, 256),
                    load_frag(Cs + ni*16*64 + k, 64), o);
    __syncthreads();
  }
  int n = ni*16 + c_col(lane);                 // dh
  int mb = mi*16 + c_row0(lane);
#pragma unroll
  for (int r = 0; r < 8; ++r) {
    int m = n0 + mb + r;
    size_t off = ((size_t)b*NSEQ + m)*DMODEL + h*DH + n;
    ao[off] += o[r] * rowinv[mb + r];
  }
}

// ---------------------------------------------------------------------------
// 12. f32 -> bf16 convert
// ---------------------------------------------------------------------------
__global__ __launch_bounds__(256)
void cvt_kernel(const float* __restrict__ src, bf16* __restrict__ dst,
                size_t total) {
  size_t i = (size_t)blockIdx.x*256 + threadIdx.x;
  if (i < total) dst[i] = (bf16)src[i];
}

// ---------------------------------------------------------------------------
// 13. Output projection GEMM + bias + residual (f32 out).
// ---------------------------------------------------------------------------
__global__ __launch_bounds__(256)
void outproj_kernel(const bf16* __restrict__ A, const bf16* __restrict__ Bt,
                    const float* __restrict__ bias, const float* __restrict__ xr,
                    float* __restrict__ out) {
  int wv = threadIdx.x >> 5, lane = threadIdx.x & 31;
#pragma unroll
  for (int s = 0; s < 2; ++s) {
    int t  = wv*2 + s;
    int m0 = blockIdx.y*64 + (t>>2)*16;
    int n0 = blockIdx.x*64 + (t&3)*16;
    f32x8 acc = {};
    for (int k = 0; k < DMODEL; k += 32)
      acc = wmma_bf16(load_frag(A  + (size_t)m0*DMODEL + k, DMODEL),
                      load_frag(Bt + (size_t)n0*DMODEL + k, DMODEL), acc);
    int n = n0 + c_col(lane), mb = m0 + c_row0(lane);
#pragma unroll
    for (int r = 0; r < 8; ++r) {
      size_t off = (size_t)(mb+r)*DMODEL + n;
      out[off] = acc[r] + bias[n] + xr[off];
    }
  }
}

// ---------------------------------------------------------------------------
// Host orchestration
// ---------------------------------------------------------------------------
extern "C" void kernel_launch(void* const* d_in, const int* in_sizes, int n_in,
                              void* d_out, int out_size, void* d_ws,
                              size_t ws_size, hipStream_t stream) {
  (void)in_sizes; (void)n_in; (void)out_size; (void)ws_size;
  const float* x     = (const float*)d_in[0];
  const float* gamma = (const float*)d_in[1];
  const float* beta  = (const float*)d_in[2];
  const float* w_qkv = (const float*)d_in[3];
  const float* rker  = (const float*)d_in[4];
  const float* w_out = (const float*)d_in[5];
  const float* b_out = (const float*)d_in[6];
  float* out = (float*)d_out;

  char* ws = (char*)d_ws;
  size_t off = 0;
  auto alloc = [&](size_t bytes) -> char* {
    char* p = ws + off;
    off = (off + bytes + 255) & ~(size_t)255;
    return p;
  };
  bf16* xn    = (bf16*)alloc((size_t)ROWS*DMODEL*2);
  bf16* qb    = (bf16*)alloc((size_t)BH*NSEQ*DH*2);
  bf16* kb    = (bf16*)alloc((size_t)BH*NSEQ*DH*2);
  bf16* vb    = (bf16*)alloc((size_t)BH*NSEQ*DH*2);
  bf16* wqkvT = (bf16*)alloc((size_t)1536*DMODEL*2);
  bf16* woutT = (bf16*)alloc((size_t)DMODEL*DMODEL*2);
  bf16* ql    = (bf16*)alloc((size_t)BH*MLAND*DH*2);
  bf16* kl    = (bf16*)alloc((size_t)BH*MLAND*DH*2);
  float* s2   = (float*)alloc((size_t)BH*MLAND*MLAND*4);
  bf16* a2b   = (bf16*)alloc((size_t)BH*MLAND*MLAND*2);
  bf16* z     = (bf16*)alloc((size_t)BH*MLAND*MLAND*2);
  bf16* z2    = (bf16*)alloc((size_t)BH*MLAND*MLAND*2);
  bf16* tT    = (bf16*)alloc((size_t)BH*MLAND*MLAND*2);
  bf16* xzb   = (bf16*)alloc((size_t)BH*MLAND*MLAND*2);
  bf16* tb    = (bf16*)alloc((size_t)BH*MLAND*MLAND*2);
  bf16* avT   = (bf16*)alloc((size_t)BH*DH*MLAND*2);
  bf16* Wt    = (bf16*)alloc((size_t)BH*DH*MLAND*2);
  float* ao   = (float*)alloc((size_t)ROWS*DMODEL*4);
  bf16* aob   = (bf16*)alloc((size_t)ROWS*DMODEL*2);
  float* perbh = (float*)alloc(BH*4);
  float* denom = (float*)alloc(256);

  const size_t sMM = (size_t)MLAND*MLAND;     // 65536
  const size_t sLD = (size_t)MLAND*DH;        // 16384

  ln_kernel<<<ROWS, 128, 0, stream>>>(x, gamma, beta, xn);
  cvtT_kernel<<<(DMODEL*1536+255)/256, 256, 0, stream>>>(w_qkv, wqkvT, DMODEL, 1536);
  cvtT_kernel<<<(DMODEL*DMODEL+255)/256, 256, 0, stream>>>(w_out, woutT, DMODEL, DMODEL);
  qkv_kernel<<<dim3(1536/64, ROWS/64), 256, 0, stream>>>(xn, wqkvT, qb, kb, vb);
  landmark_kernel<<<(BH*MLAND*DH)/256, 256, 0, stream>>>(qb, ql);
  landmark_kernel<<<(BH*MLAND*DH)/256, 256, 0, stream>>>(kb, kl);
  gemm_kernel<<<dim3(4, 4, BH), 256, 0, stream>>>(
      ql, sLD, DH, kl, sLD, DH, DH,
      1.f, 0.f, nullptr, s2, sMM, MLAND, 0, 0.f, 0.f, nullptr);
  softmax256_kernel<<<BH*MLAND, 32, 0, stream>>>(s2, a2b);
  colsummax_kernel<<<BH, 256, 0, stream>>>(a2b, perbh);
  reducemax_kernel<<<1, 32, 0, stream>>>(perbh, denom);
  initz_kernel<<<(BH*sMM)/256, 256, 0, stream>>>(a2b, denom, z);
  bf16* zc = z; bf16* zn = z2;
  for (int it = 0; it < 6; ++it) {
    trans256_kernel<<<(BH*sMM)/256, 256, 0, stream>>>(zc, tT);
    gemm_kernel<<<dim3(4, 4, BH), 256, 0, stream>>>(      // xz, t1 = 7I - xz
        a2b, sMM, MLAND, tT, sMM, MLAND, MLAND,
        1.f, 0.f, xzb, nullptr, sMM, MLAND, 0, -1.f, 7.f, tb);
    trans256_kernel<<<(BH*sMM)/256, 256, 0, stream>>>(tb, tT);
    gemm_kernel<<<dim3(4, 4, BH), 256, 0, stream>>>(      // t2 = 15I - xz t1
        xzb, sMM, MLAND, tT, sMM, MLAND, MLAND,
        -1.f, 15.f, tb, nullptr, sMM, MLAND, 0, 0.f, 0.f, nullptr);
    trans256_kernel<<<(BH*sMM)/256, 256, 0, stream>>>(tb, tT);
    gemm_kernel<<<dim3(4, 4, BH), 256, 0, stream>>>(      // t3 = 13I - xz t2
        xzb, sMM, MLAND, tT, sMM, MLAND, MLAND,
        -1.f, 13.f, tb, nullptr, sMM, MLAND, 0, 0.f, 0.f, nullptr);
    trans256_kernel<<<(BH*sMM)/256, 256, 0, stream>>>(tb, tT);
    gemm_kernel<<<dim3(4, 4, BH), 256, 0, stream>>>(      // z' = 0.25 z t3
        zc, sMM, MLAND, tT, sMM, MLAND, MLAND,
        0.25f, 0.f, zn, nullptr, sMM, MLAND, 0, 0.f, 0.f, nullptr);
    bf16* tmp = zc; zc = zn; zn = tmp;
  }
  attn3v_kernel<<<dim3(MLAND/64, BH), 256, 0, stream>>>(ql, kb, vb, avT);
  gemm_kernel<<<dim3(1, 4, BH), 256, 0, stream>>>(        // W^T = (z av)^T
      zc, sMM, MLAND, avT, sLD, MLAND, MLAND,
      1.f, 0.f, Wt, nullptr, sLD, MLAND, 1, 0.f, 0.f, nullptr);
  conv_kernel<<<(BH*NSEQ*DH)/256, 256, 0, stream>>>(vb, rker, ao);
  attn1_kernel<<<dim3(NSEQ/32, BH), 256, 0, stream>>>(qb, kl, Wt, ao);
  cvt_kernel<<<((size_t)ROWS*DMODEL)/256, 256, 0, stream>>>(ao, aob, (size_t)ROWS*DMODEL);
  outproj_kernel<<<dim3(DMODEL/64, ROWS/64), 256, 0, stream>>>(aob, woutT, b_out, x, out);
}
